// MatrixKAN_82927228551176
// MI455X (gfx1250) — compile-verified, matrix operand order checked by reading, not run on hardware
//
#include <hip/hip_runtime.h>

typedef __attribute__((ext_vector_type(16))) _Float16 v16h;
typedef __attribute__((ext_vector_type(8)))  _Float16 v8h;
typedef __attribute__((ext_vector_type(4)))  _Float16 v4h;
typedef __attribute__((ext_vector_type(8)))  float    v8f;
typedef _Float16 half_t;

#define BATCH 256
#define GRIDN 64
#define KAN_EPS 0.1f
#define BT 128                       // batch rows per block (8 WMMA M-tiles)
#define MT 8                         // M-tiles per block
#define BROW_BYTES 272               // 64 f32 + 16B pad (bank rotation)
#define AROW_HALVES 72               // 64 halves + pad

__device__ __forceinline__ unsigned int hbits(half_t h) {
    return (unsigned int)__builtin_bit_cast(unsigned short, h);
}

// ---------------------------------------------------------------------------
// Phase 0: reduce split-K partials of the previous layer and emit per-(b,i)
// spline window index + basis-weighted polynomial coefficients (f16).
// ---------------------------------------------------------------------------
__global__ void kan_coeff(const float* __restrict__ h, int nparts, int pstride,
                          const float* __restrict__ M, int IN,
                          int* __restrict__ idxArr, half_t* __restrict__ pArr)
{
    int tid = blockIdx.x * blockDim.x + threadIdx.x;
    if (tid >= BATCH * IN) return;

    float x = 0.0f;
    for (int j = 0; j < nparts; ++j) x += h[(size_t)j * pstride + tid];

    x = fminf(fmaxf(x, 0.0f), 1.0f - KAN_EPS);
    float pos = x * (float)GRIDN;
    int   idx = (int)pos;
    idx = min(max(idx, 0), GRIDN - 4);
    float t = pos - floorf(pos);
    float pb1 = t, pb2 = t * t, pb3 = t * t * t;

    idxArr[tid] = idx;
#pragma unroll
    for (int k = 0; k < 4; ++k) {
        float acc = M[0 * 4 + k] + pb1 * M[1 * 4 + k]
                  + pb2 * M[2 * 4 + k] + pb3 * M[3 * 4 + k];
        pArr[tid * 4 + k] = (half_t)acc;
    }
}

// Final split-K reduction of the last layer into d_out.
__global__ void kan_reduce(const float* __restrict__ p, int nparts, int pstride,
                           float* __restrict__ o, int n)
{
    int tid = blockIdx.x * blockDim.x + threadIdx.x;
    if (tid >= n) return;
    float s = 0.0f;
    for (int j = 0; j < nparts; ++j) s += p[(size_t)j * pstride + tid];
    o[tid] = s;
}

// ---------------------------------------------------------------------------
// Sparse-coefficient GEMM: out[b,o] = sum_{i,g} C[b,i*64+g] * W[i,o,g]
// Block = 4 wave32 = 128(b) x 64(o) tile, split-K over blockIdx.z.
// Per i (software-pipelined, double-buffered, 1 barrier/iter):
//   - B tile W[i, o0..o0+63, 0..63] -> LDS via global_load_async_to_lds_b128
//     (GVS form: SGPR base + per-thread VGPR offset), rows padded to 272B.
//   - A tile: each thread zeroes one 64-half row and writes the 4-coefficient
//     window as three ALIGNED b32 stores at (idx & ~1).
//   - 16 v_wmma_f32_16x16x32_f16 per wave per i (one B frag feeds 8 M-tiles).
// ---------------------------------------------------------------------------
__global__ void __launch_bounds__(128)
kan_spline_gemm(const float* __restrict__ W, const int* __restrict__ idxArr,
                const half_t* __restrict__ pArr, float* __restrict__ out,
                int IN, int OUT, int nsplit)
{
    __shared__ __align__(16) unsigned char braw[2][64 * BROW_BYTES];     // 2 x 17KB
    __shared__ __align__(16) half_t atile[2][BT * AROW_HALVES];          // 2 x 18KB

    const int tid  = threadIdx.x;
    const int lane = tid & 31;
    const int wave = tid >> 5;
    const int o0   = blockIdx.x * 64;
    const int b0   = blockIdx.y * BT;
    const int seg  = IN / nsplit;
    const int i0   = blockIdx.z * seg;
    const int lg   = lane >> 4;
    const int bn   = lane & 15;

    out += (size_t)blockIdx.z * BATCH * OUT;

    // ---- async-copy geometry: 8 x 16B per thread per i (16KB B tile) ----
    const float* sbase = W + ((size_t)i0 * OUT + o0) * GRIDN;   // uniform (SGPR)
    unsigned int voffr[8], ldso[8];
#pragma unroll
    for (int r = 0; r < 8; ++r) {
        int lin = (tid + r * 128) * 16;
        voffr[r] = (unsigned int)lin;
        ldso[r]  = (unsigned int)((lin >> 8) * BROW_BYTES + (lin & 255));
    }
    unsigned int bbase[2] = { (unsigned int)(uintptr_t)&braw[0][0],
                              (unsigned int)(uintptr_t)&braw[1][0] };

    // ---- A staging: thread t owns batch row b0+t ----
    int aidx = 0;
    v4h apv  = {};

    auto issueB = [&](int buf) {
#pragma unroll
        for (int r = 0; r < 8; ++r) {
            unsigned int l = bbase[buf] + ldso[r];
            asm volatile("global_load_async_to_lds_b128 %0, %1, %2"
                         :: "v"(l), "v"(voffr[r]), "s"(sbase) : "memory");
        }
        sbase += (size_t)OUT * GRIDN;                 // scalar advance per i
    };
    auto loadA = [&](int k) {
        int ii = (b0 + tid) * IN + (i0 + k);
        aidx = idxArr[ii];
        apv  = *(const v4h*)(pArr + (size_t)ii * 4);
    };
    auto storeA = [&](int buf) {
        half_t* rowp = &atile[buf][tid * AROW_HALVES];
        v8h z = {};
#pragma unroll
        for (int s = 0; s < 8; ++s) *(v8h*)(rowp + s * 8) = z;
        // aligned 3 x b32 window write at (idx & ~1); neighbors are zero anyway
        int e = aidx & ~1;
        unsigned int p0 = hbits(apv[0]), p1 = hbits(apv[1]);
        unsigned int p2 = hbits(apv[2]), p3 = hbits(apv[3]);
        bool odd = (aidx & 1) != 0;
        unsigned int d0 = odd ? (p0 << 16)        : (p0 | (p1 << 16));
        unsigned int d1 = odd ? (p1 | (p2 << 16)) : (p2 | (p3 << 16));
        unsigned int d2 = odd ? p3                : 0u;
        unsigned int* wp = (unsigned int*)(rowp + e);
        wp[0] = d0; wp[1] = d1; wp[2] = d2;
    };

    v8f acc[MT];
#pragma unroll
    for (int mt = 0; mt < MT; ++mt) acc[mt] = (v8f){};

    // ---- pipeline prologue ----
    issueB(0);
    loadA(0);

    for (int k = 0; k < seg; ++k) {
        const int buf = k & 1;

        storeA(buf);                                        // A(k) -> LDS
        asm volatile("s_wait_asynccnt 0x0" ::: "memory");   // B(k) copies done
        __syncthreads();                                    // all waves visible

        if (k + 1 < seg) {                                  // overlap with WMMA
            issueB(buf ^ 1);
            loadA(k + 1);
        }

        const unsigned char* browp = &braw[buf][(wave * 16 + bn) * BROW_BYTES];
        const half_t* abase = &atile[buf][bn * AROW_HALVES];
#pragma unroll
        for (int gc = 0; gc < 2; ++gc) {
            // B fragment: two contiguous 8-float runs, cvt to f16
            const float* rp = (const float*)(browp) + gc * 32 + 8 * lg;
            float4 q0 = ((const float4*)rp)[0];
            float4 q1 = ((const float4*)rp)[1];
            float4 q2 = ((const float4*)(rp + 16))[0];
            float4 q3 = ((const float4*)(rp + 16))[1];
            v16h b;
            b[0]=(half_t)q0.x;  b[1]=(half_t)q0.y;  b[2]=(half_t)q0.z;  b[3]=(half_t)q0.w;
            b[4]=(half_t)q1.x;  b[5]=(half_t)q1.y;  b[6]=(half_t)q1.z;  b[7]=(half_t)q1.w;
            b[8]=(half_t)q2.x;  b[9]=(half_t)q2.y;  b[10]=(half_t)q2.z; b[11]=(half_t)q2.w;
            b[12]=(half_t)q3.x; b[13]=(half_t)q3.y; b[14]=(half_t)q3.z; b[15]=(half_t)q3.w;

            // one B fragment feeds 8 A fragments (8 M-tiles)
#pragma unroll
            for (int mt = 0; mt < MT; ++mt) {
                const half_t* arp = abase + mt * (16 * AROW_HALVES) + gc * 32 + 8 * lg;
                v8h a0 = *(const v8h*)(arp);
                v8h a1 = *(const v8h*)(arp + 16);
                v16h a;
#pragma unroll
                for (int j = 0; j < 8; ++j) { a[j] = a0[j]; a[j + 8] = a1[j]; }
                acc[mt] = __builtin_amdgcn_wmma_f32_16x16x32_f16(
                              false, a, false, b, (short)0, acc[mt], false, false);
            }
        }
    }

    // ---- store C tiles: VGPR r -> M = r + 8*(lane>=16), N = lane&15 ----
    const int mo = (lane >> 4) * 8;
#pragma unroll
    for (int mt = 0; mt < MT; ++mt)
#pragma unroll
        for (int r = 0; r < 8; ++r)
            out[(size_t)(b0 + mt * 16 + mo + r) * OUT + (o0 + wave * 16 + bn)] = acc[mt][r];
}

// ---------------------------------------------------------------------------
extern "C" void kernel_launch(void* const* d_in, const int* in_sizes, int n_in,
                              void* d_out, int out_size, void* d_ws, size_t ws_size,
                              hipStream_t stream)
{
    const float* x  = (const float*)d_in[0];
    const float* W1 = (const float*)d_in[1];
    const float* W2 = (const float*)d_in[2];
    const float* W3 = (const float*)d_in[3];
    const float* M  = (const float*)d_in[4];

    // split-K factor chosen from available scratch (deterministic)
    const int ks = (ws_size >= (10ull << 20)) ? 8 : 2;

    const size_t hN = (size_t)BATCH * 512;          // hidden activations
    const size_t fN = (size_t)BATCH * 256;          // final activations
    char* ws = (char*)d_ws;
    float*  h1p = (float*)(ws);                                   // ks * 512KB
    float*  h2p = (float*)(ws + ks * hN * sizeof(float));         // ks * 512KB
    char*   tail = ws + 2 * ks * hN * sizeof(float);
    int*    idx = (int*)tail;                                     // 512KB
    half_t* pc  = (half_t*)(tail + (512u << 10));                 // 1MB
    float*  fp  = h1p;                                            // reuse for final partials
    float*  o   = (float*)d_out;

    // layer 1: IN=256, OUT=512
    kan_coeff<<<dim3((BATCH * 256) / 256), dim3(256), 0, stream>>>(x, 1, 0, M, 256, idx, pc);
    kan_spline_gemm<<<dim3(512 / 64, BATCH / BT, ks), dim3(128), 0, stream>>>(W1, idx, pc, h1p, 256, 512, ks);
    // layer 2: IN=512, OUT=512  (coeff folds the split-K reduction of h1)
    kan_coeff<<<dim3((BATCH * 512) / 256), dim3(256), 0, stream>>>(h1p, ks, (int)hN, M, 512, idx, pc);
    kan_spline_gemm<<<dim3(512 / 64, BATCH / BT, ks), dim3(128), 0, stream>>>(W2, idx, pc, h2p, 512, 512, ks);
    // layer 3: IN=512, OUT=256
    kan_coeff<<<dim3((BATCH * 512) / 256), dim3(256), 0, stream>>>(h2p, ks, (int)hN, M, 512, idx, pc);
    kan_spline_gemm<<<dim3(256 / 64, BATCH / BT, ks), dim3(128), 0, stream>>>(W3, idx, pc, fp, 512, 256, ks);
    kan_reduce<<<dim3((BATCH * 256) / 256), dim3(256), 0, stream>>>(fp, ks, (int)fN, o, (int)fN);
}
